// StrideMoEOCR_74131135529162
// MI455X (gfx1250) — matrix operand order repaired and checked by reading, not compiled, not run on hardware
//
#include <hip/hip_runtime.h>
#include <hip/hip_bf16.h>
#include <math.h>

// ---------------------------------------------------------------------------
// MI455X / gfx1250 implementation.
// bf16 WMMA (v_wmma_f32_16x16x32_bf16, f32 accum) for all GEMMs.
// LDS tiles stored in WMMA fragment order -> every fragment load is
// 2x ds_load_b128; A-operands loaded as 2x global_load_b128 directly into
// fragment registers. GEMM: 256x64 workgroup tile, 32x64 per wave (each B
// fragment feeds 2 WMMAs), double-buffered B tile, fragments preloaded so
// WMMAs issue back-to-back.
// ---------------------------------------------------------------------------

typedef __attribute__((ext_vector_type(16))) __bf16 v16bf;
typedef __attribute__((ext_vector_type(8)))  __bf16 v8bf;
typedef __attribute__((ext_vector_type(8)))  float  v8f;

static __device__ __forceinline__ __bf16 f2bf(float f) {
  union { float f; unsigned u; } v; v.f = f;
  unsigned r = v.u + 0x7FFFu + ((v.u >> 16) & 1u);   // round-to-nearest-even
  unsigned short h = (unsigned short)(r >> 16);
  return __builtin_bit_cast(__bf16, h);
}
static __device__ __forceinline__ float bf2f(__bf16 b) {
  unsigned short h = __builtin_bit_cast(unsigned short, b);
  unsigned u = ((unsigned)h) << 16;
  return __builtin_bit_cast(float, u);
}
static __device__ __forceinline__ float gelu_exact(float x) {
  return 0.5f * x * (1.0f + erff(x * 0.70710678118654752f));
}
static __device__ __forceinline__ v16bf cat16(v8bf lo, v8bf hi) {
  v16bf a;
  #pragma unroll
  for (int i = 0; i < 8; ++i) { a[i] = lo[i]; a[8 + i] = hi[i]; }
  return a;
}

// ---------------------------------------------------------------------------
// Direct 3x3 conv + exact GELU (fp32). One thread per output element.
// ---------------------------------------------------------------------------
__global__ __launch_bounds__(256) void conv3x3_gelu_kernel(
    const float* __restrict__ in, const float* __restrict__ w,
    const float* __restrict__ bias, float* __restrict__ out,
    int Bn, int Cin, int Hi, int Wi, int Cout, int Ho, int Wo, int stride)
{
  size_t idx = (size_t)blockIdx.x * 256 + threadIdx.x;
  size_t total = (size_t)Bn * Cout * Ho * Wo;
  if (idx >= total) return;
  int wo = (int)(idx % Wo); size_t t = idx / Wo;
  int ho = (int)(t % Ho);   t /= Ho;
  int co = (int)(t % Cout); int b = (int)(t / Cout);
  float acc = bias[co];
  int hi0 = ho * stride - 1, wi0 = wo * stride - 1;
  for (int ci = 0; ci < Cin; ++ci) {
    const float* ip = in + ((size_t)(b * Cin + ci) * Hi) * Wi;
    const float* wp = w + ((size_t)(co * Cin + ci)) * 9;
    #pragma unroll
    for (int kh = 0; kh < 3; ++kh) {
      int hh = hi0 + kh;
      if ((unsigned)hh >= (unsigned)Hi) continue;
      #pragma unroll
      for (int kw = 0; kw < 3; ++kw) {
        int ww = wi0 + kw;
        if ((unsigned)ww >= (unsigned)Wi) continue;
        acc += ip[(size_t)hh * Wi + ww] * wp[kh * 3 + kw];
      }
    }
  }
  out[idx] = gelu_exact(acc);
}

// ---------------------------------------------------------------------------
// Mean-pool over H' (8) + transpose + sinusoidal positional encoding.
// f4: [32, 256, 8, 128] -> h: [4096, 256]
// ---------------------------------------------------------------------------
__global__ __launch_bounds__(256) void pool_pos_kernel(
    const float* __restrict__ f4, float* __restrict__ h)
{
  int idx = blockIdx.x * 256 + threadIdx.x;        // 4096*256 total
  int d = idx & 255; int tok = idx >> 8;
  int t = tok & 127; int b = tok >> 7;
  const float* p = f4 + (((size_t)b * 256 + d) * 8) * 128 + t;
  float s = 0.f;
  #pragma unroll
  for (int hh = 0; hh < 8; ++hh) s += p[(size_t)hh * 128];
  s *= 0.125f;
  float freq = __expf(-9.210340371976184f * (float)(d & ~1) / 256.0f);
  float pe = (d & 1) ? cosf((float)t * freq) : sinf((float)t * freq);
  h[idx] = s + pe;
}

// ---------------------------------------------------------------------------
// LayerNorm over D=256. One block per row. Writes f32 (optional) + bf16.
// ---------------------------------------------------------------------------
__global__ __launch_bounds__(256) void ln_kernel(
    const float* __restrict__ x, const float* __restrict__ g,
    const float* __restrict__ be, float* __restrict__ yf,
    __bf16* __restrict__ ybf)
{
  int row = blockIdx.x, tid = threadIdx.x;
  float v = x[(size_t)row * 256 + tid];
  __shared__ float red[256];
  red[tid] = v; __syncthreads();
  for (int off = 128; off; off >>= 1) { if (tid < off) red[tid] += red[tid + off]; __syncthreads(); }
  float mean = red[0] * (1.0f / 256.0f);
  __syncthreads();
  float d = v - mean;
  red[tid] = d * d; __syncthreads();
  for (int off = 128; off; off >>= 1) { if (tid < off) red[tid] += red[tid + off]; __syncthreads(); }
  float var = red[0] * (1.0f / 256.0f);
  float o = d * rsqrtf(var + 1e-5f) * g[tid] + be[tid];
  if (yf) yf[(size_t)row * 256 + tid] = o;
  ybf[(size_t)row * 256 + tid] = f2bf(o);
}

// ---------------------------------------------------------------------------
// WMMA-bf16 GEMM:  C[M,N] = A[M,K](bf16) * Bw[N,K]^T(fp32->bf16) + bias
// 256 thr / 8 waves; workgroup tile 256(M) x 64(N) x 32(K); 32x64 per wave.
// A fragments: two 16B global runs per lane (k-pattern [kadd,kadd+8) u
// [kadd+16,kadd+24)). B tile double-buffered in LDS fragment order; the 4
// fragments are preloaded into registers, then 8 WMMAs issue back-to-back
// (each B fragment reused by both A row-fragments -> half the LDS traffic).
// Requires M%256==0, N%64==0, K%32==0 (all shapes here satisfy this).
// Epilogue modes: 0 store(+bias), 1 gelu->bf16, 2 Cf+=, 3 Cf+=rowscale*().
// ---------------------------------------------------------------------------
__global__ __launch_bounds__(256) void gemm_bf16_kernel(
    const __bf16* __restrict__ A, const float* __restrict__ Bw,
    const float* __restrict__ bias, float* __restrict__ Cf,
    __bf16* __restrict__ Cb, const float* __restrict__ rowscale, int rsStride,
    int M, int N, int K, int mode)
{
  __shared__ __align__(32) __bf16 Bs[2][4 * 32 * 16];
  int tid = threadIdx.x;
  int wv = tid >> 5, lane = tid & 31;
  int mblk = blockIdx.y * 256, nblk = blockIdx.x * 64;

  v8f acc[2][4];
  #pragma unroll
  for (int g = 0; g < 2; ++g)
    #pragma unroll
    for (int nt = 0; nt < 4; ++nt)
      acc[g][nt] = v8f{0.f,0.f,0.f,0.f,0.f,0.f,0.f,0.f};

  // B staging: thread -> weight row n, 8 consecutive k. Fragment slot:
  // nt=n>>4, slane=((k0&8)?16:0)+(n&15), e=(k0&16)>>1 -> contiguous 8 elems (16B).
  int nb = tid >> 2, k0 = (tid & 3) * 8;
  int sslot = ((nb >> 4) * 32 + ((k0 & 8) ? 16 : 0) + (nb & 15)) * 16 + ((k0 & 16) >> 1);
  const float* brow = Bw + (size_t)(nblk + nb) * K + k0;

  // Two A row-fragments per wave (rows wv*32+0..15 and +16..31).
  int kadd = (lane & 16) ? 8 : 0;
  const __bf16* abase0 = A + (size_t)(mblk + wv * 32 + (lane & 15)) * K + kadd;
  const __bf16* abase1 = abase0 + (size_t)16 * K;

  v16bf a_cur[2], a_next[2];
  a_cur[0] = cat16(*(const v8bf*)(abase0), *(const v8bf*)(abase0 + 16));
  a_cur[1] = cat16(*(const v8bf*)(abase1), *(const v8bf*)(abase1 + 16));
  { v8bf t;
    #pragma unroll
    for (int i = 0; i < 8; ++i) t[i] = f2bf(brow[i]);
    *(v8bf*)&Bs[0][sslot] = t; }
  __syncthreads();

  int buf = 0;
  for (int kb = 0; kb < K; kb += 32) {
    bool more = (kb + 32) < K;
    if (more) {
      a_next[0] = cat16(*(const v8bf*)(abase0 + kb + 32),
                        *(const v8bf*)(abase0 + kb + 48));
      a_next[1] = cat16(*(const v8bf*)(abase1 + kb + 32),
                        *(const v8bf*)(abase1 + kb + 48));
      const float* bp = brow + kb + 32;
      v8bf t;
      #pragma unroll
      for (int i = 0; i < 8; ++i) t[i] = f2bf(bp[i]);
      *(v8bf*)&Bs[buf ^ 1][sslot] = t;
      if (kb + 64 < K) __builtin_prefetch(brow + kb + 64, 0, 1);
    }
    // Preload all 4 B fragments, then issue 8 WMMAs back-to-back.
    v16bf bfr[4];
    #pragma unroll
    for (int nt = 0; nt < 4; ++nt)
      bfr[nt] = *(const v16bf*)&Bs[buf][(nt * 32 + lane) * 16];
    #pragma unroll
    for (int nt = 0; nt < 4; ++nt) {
      acc[0][nt] = __builtin_amdgcn_wmma_f32_16x16x32_bf16(
          false, a_cur[0], false, bfr[nt], (short)0, acc[0][nt], false, false);
      acc[1][nt] = __builtin_amdgcn_wmma_f32_16x16x32_bf16(
          false, a_cur[1], false, bfr[nt], (short)0, acc[1][nt], false, false);
    }
    if (more) { __syncthreads(); a_cur[0] = a_next[0]; a_cur[1] = a_next[1]; buf ^= 1; }
  }

  // Epilogue. C layout: VGPR r -> M=r (lanes 0-15) / r+8 (lanes 16-31), N=lane%16.
  #pragma unroll
  for (int g = 0; g < 2; ++g) {
    #pragma unroll
    for (int nt = 0; nt < 4; ++nt) {
      #pragma unroll
      for (int r = 0; r < 8; ++r) {
        int rowl = wv * 32 + g * 16 + ((lane & 16) ? (r + 8) : r);
        int coll = nt * 16 + (lane & 15);
        size_t m = (size_t)(mblk + rowl); int n = nblk + coll;
        float val = acc[g][nt][r] + (bias ? bias[n] : 0.0f);
        size_t off = m * (size_t)N + n;
        if (mode == 0)      { if (Cf) Cf[off] = val; if (Cb) Cb[off] = f2bf(val); }
        else if (mode == 1) { Cb[off] = f2bf(gelu_exact(val)); }
        else if (mode == 2) { Cf[off] += val; }
        else                { Cf[off] += rowscale[m * rsStride] * val; }
      }
    }
  }
}

// ---------------------------------------------------------------------------
// Fused attention: one WG (256 thr / 8 waves) per (batch, head).
// T=128, hd=32, scale=1/sqrt(32).
// Q fragments from global (2x 16B). K and V tiles staged in LDS in WMMA
// fragment order (vector reads). P kept bf16 in LDS, row stride 136 so the
// A-fragment 16B runs stay aligned.
// ---------------------------------------------------------------------------
__global__ __launch_bounds__(256) void attn_kernel(
    const __bf16* __restrict__ qkv, __bf16* __restrict__ attnout)
{
  __shared__ __align__(32) __bf16 Ksf[8 * 32 * 16];   // S-phase B fragments
  __shared__ __align__(32) __bf16 Vsf[8 * 32 * 16];   // O-phase B fragments
  __shared__ __align__(32) __bf16 Ps[128][136];       // scores then probs

  int bh = blockIdx.x; int b = bh >> 3; int hd = bh & 7;
  int tid = threadIdx.x;
  int wv = tid >> 5, lane = tid & 31;
  int kadd = (lane & 16) ? 16 : 0;

  // ---- stage K (fragment order, fully contiguous both sides) ----
  {
    int n = tid >> 1, khalf = (tid & 1);               // khalf: k block 0..15 / 16..31
    const __bf16* src = qkv + (size_t)(b * 128 + n) * 768 + 256 + hd * 32 + khalf * 16;
    v16bf t = *(const v16bf*)src;
    *(v16bf*)&Ksf[((n >> 4) * 32 + khalf * 16 + (n & 15)) * 16] = t;
  }
  // ---- stage V (vector global read, scatter LDS write; read back 8x) ----
  {
    int k = tid >> 1, hofs = (tid & 1) * 16;           // k = token, hofs = n block
    const __bf16* src = qkv + (size_t)(b * 128 + k) * 768 + 512 + hd * 32 + hofs;
    v16bf t = *(const v16bf*)src;
    int frag = (k >> 5) * 2 + (hofs >> 4);
    int lbase = (k & 16) ? 16 : 0;
    int e = k & 15;
    #pragma unroll
    for (int i = 0; i < 16; ++i)
      Vsf[(frag * 32 + lbase + i) * 16 + e] = t[i];
  }
  __syncthreads();

  // ---- S = scale * Q K^T ----
  {
    const __bf16* qbase =
        qkv + (size_t)(b * 128 + wv * 16 + (lane & 15)) * 768 + hd * 32 + kadd;
    v16bf a = cat16(*(const v8bf*)qbase, *(const v8bf*)(qbase + 8));
    #pragma unroll
    for (int nt = 0; nt < 8; ++nt) {
      v16bf bfr = *(const v16bf*)&Ksf[(nt * 32 + lane) * 16];
      v8f c = {0.f,0.f,0.f,0.f,0.f,0.f,0.f,0.f};
      c = __builtin_amdgcn_wmma_f32_16x16x32_bf16(
          false, a, false, bfr, (short)0, c, false, false);
      #pragma unroll
      for (int r = 0; r < 8; ++r) {
        int rowl = wv * 16 + ((lane & 16) ? (r + 8) : r);
        Ps[rowl][nt * 16 + (lane & 15)] = f2bf(c[r] * 0.17677669529663689f);
      }
    }
  }
  __syncthreads();

  // ---- row softmax (threads 0..127; recompute exp, no register array) ----
  if (tid < 128) {
    float mx = -3.0e38f;
    for (int j = 0; j < 128; ++j) mx = fmaxf(mx, bf2f(Ps[tid][j]));
    float s = 0.f;
    for (int j = 0; j < 128; ++j) s += __expf(bf2f(Ps[tid][j]) - mx);
    float inv = 1.0f / s;
    for (int j = 0; j < 128; ++j)
      Ps[tid][j] = f2bf(__expf(bf2f(Ps[tid][j]) - mx) * inv);
  }
  __syncthreads();

  // ---- O = P V (preload fragments, then back-to-back WMMAs) ----
  {
    int mrow = wv * 16 + (lane & 15);
    v16bf pa[4];
    #pragma unroll
    for (int kk = 0; kk < 4; ++kk)
      pa[kk] = cat16(*(const v8bf*)&Ps[mrow][kk * 32 + kadd],
                     *(const v8bf*)&Ps[mrow][kk * 32 + kadd + 8]);
    #pragma unroll
    for (int nt = 0; nt < 2; ++nt) {
      v16bf vb[4];
      #pragma unroll
      for (int kk = 0; kk < 4; ++kk)
        vb[kk] = *(const v16bf*)&Vsf[((kk * 2 + nt) * 32 + lane) * 16];
      v8f c = {0.f,0.f,0.f,0.f,0.f,0.f,0.f,0.f};
      #pragma unroll
      for (int kk = 0; kk < 4; ++kk)
        c = __builtin_amdgcn_wmma_f32_16x16x32_bf16(
            false, pa[kk], false, vb[kk], (short)0, c, false, false);
      #pragma unroll
      for (int r = 0; r < 8; ++r) {
        int rowl = wv * 16 + ((lane & 16) ? (r + 8) : r);
        int col  = nt * 16 + (lane & 15);
        attnout[(size_t)(b * 128 + rowl) * 256 + hd * 32 + col] = f2bf(c[r]);
      }
    }
  }
}

// ---------------------------------------------------------------------------
// Router: one block per token -> softmax probs [tok,8] + top-2 combine [tok,8].
// ---------------------------------------------------------------------------
__global__ __launch_bounds__(256) void router_kernel(
    const float* __restrict__ yf, const float* __restrict__ rw,
    const float* __restrict__ rb, float* __restrict__ probs,
    float* __restrict__ comb)
{
  int tok = blockIdx.x, tid = threadIdx.x;
  int e = tid >> 5, lane = tid & 31;
  const float* y = yf + (size_t)tok * 256;
  const float* w = rw + (size_t)e * 256;
  float s = 0.f;
  for (int k = lane; k < 256; k += 32) s += y[k] * w[k];
  __shared__ float red[256];
  __shared__ float logits[8];
  red[tid] = s; __syncthreads();
  for (int off = 16; off; off >>= 1) { if (lane < off) red[tid] += red[tid + off]; __syncthreads(); }
  if (lane == 0) logits[e] = red[tid] + rb[e];
  __syncthreads();
  if (tid == 0) {
    float mx = -3.0e38f;
    for (int i = 0; i < 8; ++i) mx = fmaxf(mx, logits[i]);
    float ex[8], ss = 0.f;
    for (int i = 0; i < 8; ++i) { ex[i] = __expf(logits[i] - mx); ss += ex[i]; }
    float inv = 1.0f / ss;
    for (int i = 0; i < 8; ++i) probs[(size_t)tok * 8 + i] = ex[i] * inv;
    int i1 = 0;
    for (int i = 1; i < 8; ++i) if (logits[i] > logits[i1]) i1 = i;
    int i2 = -1;
    for (int i = 0; i < 8; ++i) if (i != i1 && (i2 < 0 || logits[i] > logits[i2])) i2 = i;
    float m2 = fmaxf(logits[i1], logits[i2]);
    float e1 = __expf(logits[i1] - m2), e2 = __expf(logits[i2] - m2);
    float g = 1.0f / (e1 + e2);
    for (int i = 0; i < 8; ++i) comb[(size_t)tok * 8 + i] = 0.f;
    comb[(size_t)tok * 8 + i1] += e1 * g;
    comb[(size_t)tok * 8 + i2] += e2 * g;
  }
}

// Deterministic aux-loss reduction (single WG, fixed tree).
__global__ __launch_bounds__(256) void aux_kernel(
    const float* __restrict__ probs, float* __restrict__ auxAcc)
{
  int tid = threadIdx.x;
  int e = tid & 7, chunk = tid >> 3;
  float s = 0.f;
  for (int t = chunk * 128; t < chunk * 128 + 128; ++t) s += probs[(size_t)t * 8 + e];
  __shared__ float red[256];
  red[tid] = s; __syncthreads();
  for (int off = 128; off >= 8; off >>= 1) { if (tid < off) red[tid] += red[tid + off]; __syncthreads(); }
  if (tid == 0) {
    float aux = 0.f;
    for (int i = 0; i < 8; ++i) {
      float d = red[i] * (1.0f / 4096.0f) - 0.125f;
      aux += d * d;
    }
    auxAcc[0] += aux * (1.0f / 8.0f);
  }
}

__global__ void init_aux_kernel(float* a) { a[0] = 0.f; }
__global__ void write_aux_kernel(const float* __restrict__ a, float* __restrict__ out) {
  out[0] = a[0] * (1.0f / 6.0f);
}

// ---------------------------------------------------------------------------
// Host launch sequence.
// ---------------------------------------------------------------------------
extern "C" void kernel_launch(void* const* d_in, const int* in_sizes, int n_in,
                              void* d_out, int out_size, void* d_ws, size_t ws_size,
                              hipStream_t stream) {
  (void)in_sizes; (void)n_in; (void)out_size; (void)ws_size;
  const float* x   = (const float*)d_in[0];
  const float* cw1 = (const float*)d_in[1];  const float* cb1 = (const float*)d_in[2];
  const float* cw2 = (const float*)d_in[3];  const float* cb2 = (const float*)d_in[4];
  const float* cw3 = (const float*)d_in[5];  const float* cb3 = (const float*)d_in[6];
  const float* cw4 = (const float*)d_in[7];  const float* cb4 = (const float*)d_in[8];
  const float* ln1_g = (const float*)d_in[9];  const float* ln1_b = (const float*)d_in[10];
  const float* qkv_w = (const float*)d_in[11]; const float* qkv_b = (const float*)d_in[12];
  const float* out_w = (const float*)d_in[13]; const float* out_b = (const float*)d_in[14];
  const float* ln2_g = (const float*)d_in[15]; const float* ln2_b = (const float*)d_in[16];
  const float* rt_w  = (const float*)d_in[17]; const float* rt_b  = (const float*)d_in[18];
  const float* e_w1  = (const float*)d_in[19]; const float* e_b1  = (const float*)d_in[20];
  const float* e_w2  = (const float*)d_in[21]; const float* e_b2  = (const float*)d_in[22];
  const float* nf_g  = (const float*)d_in[23]; const float* nf_b  = (const float*)d_in[24];
  const float* hw    = (const float*)d_in[25]; const float* hb    = (const float*)d_in[26];

  char* ws = (char*)d_ws;
  // Conv intermediates (peak 201.3 MB): f1 @0 (134MB), f2 @134MB (67MB);
  // f3 reuses f1 region, f4 follows; transformer state lives past f4 (<=95MB).
  float*  f1     = (float*)(ws + 0);
  float*  f2     = (float*)(ws + 134217728);
  float*  f3     = (float*)(ws + 0);
  float*  f4     = (float*)(ws + 33554432);
  float*  h      = (float*)(ws + 67108864);   // [4096,256] f32 residual stream
  float*  yf     = (float*)(ws + 71303168);   // [4096,256] f32 (router input)
  __bf16* ybf    = (__bf16*)(ws + 75497472);  // [4096,256] bf16
  __bf16* qkvbf  = (__bf16*)(ws + 77594624);  // [4096,768] bf16
  __bf16* attnbf = (__bf16*)(ws + 83886080);  // [4096,256] bf16
  __bf16* h1bf   = (__bf16*)(ws + 85983232);  // [4096,1024] bf16
  float*  probs  = (float*)(ws + 94371840);   // [4096,8]
  float*  comb   = (float*)(ws + 94502912);   // [4096,8]
  float*  auxAcc = (float*)(ws + 94633984);   // [1]

  init_aux_kernel<<<1, 1, 0, stream>>>(auxAcc);

  // ---- conv stem (fused GELU) ----
  conv3x3_gelu_kernel<<<131072, 256, 0, stream>>>(x,  cw1, cb1, f1, 32,   1,  64, 1024,  64, 32, 512, 2);
  conv3x3_gelu_kernel<<< 65536, 256, 0, stream>>>(f1, cw2, cb2, f2, 32,  64,  32,  512, 128, 16, 256, 2);
  conv3x3_gelu_kernel<<< 32768, 256, 0, stream>>>(f2, cw3, cb3, f3, 32, 128,  16,  256, 256,  8, 128, 2);
  conv3x3_gelu_kernel<<< 32768, 256, 0, stream>>>(f3, cw4, cb4, f4, 32, 256,   8,  128, 256,  8, 128, 1);
  pool_pos_kernel<<<4096, 256, 0, stream>>>(f4, h);

  const int M = 4096;
  for (int l = 0; l < 6; ++l) {
    // --- attention block ---
    ln_kernel<<<4096, 256, 0, stream>>>(h, ln1_g + l * 256, ln1_b + l * 256, nullptr, ybf);
    gemm_bf16_kernel<<<dim3(768 / 64, M / 256), 256, 0, stream>>>(
        ybf, qkv_w + (size_t)l * 768 * 256, qkv_b + (size_t)l * 768,
        nullptr, qkvbf, nullptr, 0, M, 768, 256, 0);
    attn_kernel<<<256, 256, 0, stream>>>(qkvbf, attnbf);
    gemm_bf16_kernel<<<dim3(256 / 64, M / 256), 256, 0, stream>>>(
        attnbf, out_w + (size_t)l * 256 * 256, out_b + (size_t)l * 256,
        h, nullptr, nullptr, 0, M, 256, 256, 2);           // h += proj

    // --- MoE block (dense over all 8 experts, like the reference) ---
    ln_kernel<<<4096, 256, 0, stream>>>(h, ln2_g + l * 256, ln2_b + l * 256, yf, ybf);
    router_kernel<<<4096, 256, 0, stream>>>(yf, rt_w + (size_t)l * 8 * 256,
                                            rt_b + (size_t)l * 8, probs, comb);
    aux_kernel<<<1, 256, 0, stream>>>(probs, auxAcc);
    for (int e = 0; e < 8; ++e) {
      const float* w1 = e_w1 + ((size_t)(l * 8 + e)) * 1024 * 256;
      const float* b1 = e_b1 + ((size_t)(l * 8 + e)) * 1024;
      const float* w2 = e_w2 + ((size_t)(l * 8 + e)) * 256 * 1024;
      const float* b2 = e_b2 + ((size_t)(l * 8 + e)) * 256;
      gemm_bf16_kernel<<<dim3(1024 / 64, M / 256), 256, 0, stream>>>(
          ybf, w1, b1, nullptr, h1bf, nullptr, 0, M, 1024, 256, 1);   // gelu -> bf16
      gemm_bf16_kernel<<<dim3(256 / 64, M / 256), 256, 0, stream>>>(
          h1bf, w2, b2, h, nullptr, comb + e, 8, M, 256, 1024, 3);    // h += comb_e * (.)
    }
  }

  // --- final LN + vocab head ---
  ln_kernel<<<4096, 256, 0, stream>>>(h, nf_g, nf_b, nullptr, ybf);
  gemm_bf16_kernel<<<dim3(8000 / 64, M / 256), 256, 0, stream>>>(
      ybf, hw, hb, (float*)d_out, nullptr, nullptr, 0, M, 8000, 256, 0);
  write_aux_kernel<<<1, 1, 0, stream>>>(auxAcc, (float*)d_out + 32768000);
}
